// DecoderWithAttention_68882685493737
// MI455X (gfx1250) — matrix operand order repaired and checked by reading, not compiled, not run on hardware
//
#include <hip/hip_runtime.h>
#include <hip/hip_bf16.h>

// ---------------- problem constants (from reference) ----------------
#define Bn   64
#define Pn   196
#define En   2048
#define Dn   512
#define An   512
#define EMBn 512
#define Vn   10000
#define Tn   20
#define XW   3072          // width of x2 = EMB + E + D = 512 + 2048 + 512
#define NFCPAD 10112       // Vn padded up to multiple of 128 (79 * 128)

typedef __attribute__((ext_vector_type(16))) __bf16 v16bf;
typedef __attribute__((ext_vector_type(8)))  float  v8f;
typedef __attribute__((ext_vector_type(8)))  __bf16 bf16x8;

__device__ __forceinline__ float sigf(float x) { return 1.0f / (1.0f + __expf(-x)); }

// ---------------------------------------------------------------
// GEMM: C[M,N] = act( A[M,K](f32) * W[N,K]^T(bf16) + bias )
// block = 256 thr = 8 waves; block tile 64(M) x 128(N); K chunk 64.
// Wave (wrow,wcol) owns a 32x32 macro tile -> 2x2 WMMA tiles x 2 k-subchunks
// = 8 v_wmma_f32_16x16x32_bf16 per staging round.
// REQUIRES: M % 64 == 0, K % 64 == 0, W buffer readable up to
// ceil(N/128)*128 rows (pad weights), store guarded by n < N.
// ---------------------------------------------------------------
#define MT 64
#define NT 128
#define KC 64
#define LDK 72            // KC + 8 bf16 skew (row stride stays 16B aligned)

__global__ void gemm_bf16_wmma(const float* __restrict__ A,
                               const __bf16* __restrict__ W,
                               const float* __restrict__ bias,
                               float* __restrict__ C,
                               int M, int N, int K, int ldc, int act) {
    __shared__ __bf16 sA[MT * LDK];   // 9216 B
    __shared__ __bf16 sW[NT * LDK];   // 18432 B
    const int tid  = threadIdx.x;
    const int lane = tid & 31;
    const int wave = tid >> 5;        // 0..7
    const int wrow = wave >> 2;       // 0..1  -> 32-row slab
    const int wcol = wave & 3;        // 0..3  -> 32-col slab
    const int m0 = blockIdx.y * MT;
    const int n0 = blockIdx.x * NT;

    v8f acc[2][2] = {};

    // staging maps (uniform, no guards)
    const int ar = tid >> 2;              // 0..63   A row
    const int ac = (tid & 3) * 16;        // 0/16/32/48
    const int wr = tid >> 1;              // 0..127  W row
    const int wc = (tid & 1) * 32;        // 0/32

    const float*  Abase = A + (size_t)(m0 + ar) * K + ac;
    const __bf16* Wbase = W + (size_t)(n0 + wr) * K + wc;

    for (int kb = 0; kb < K; kb += KC) {
        // ---- stage A tile 64x64 f32 -> bf16 (16 elems / thread) ----
        {
            const float* src = Abase + kb;
            if (kb + KC < K) __builtin_prefetch(src + KC, 0, 3);
            __bf16* dst = sA + ar * LDK + ac;
#pragma unroll
            for (int q = 0; q < 2; ++q) {
                float4 f0 = *(const float4*)(src + q * 8);
                float4 f1 = *(const float4*)(src + q * 8 + 4);
                bf16x8 v;
                v[0] = (__bf16)f0.x; v[1] = (__bf16)f0.y;
                v[2] = (__bf16)f0.z; v[3] = (__bf16)f0.w;
                v[4] = (__bf16)f1.x; v[5] = (__bf16)f1.y;
                v[6] = (__bf16)f1.z; v[7] = (__bf16)f1.w;
                *(bf16x8*)(dst + q * 8) = v;
            }
        }
        // ---- stage W tile 128x64 bf16 (32 elems / thread, raw copy) ----
        {
            const uint4* src = (const uint4*)(Wbase + kb);
            uint4* dst = (uint4*)(sW + wr * LDK + wc);
#pragma unroll
            for (int q = 0; q < 4; ++q) dst[q] = src[q];
        }
        __syncthreads();

        // ---- 2 k-subchunks x 2x2 tiles = 8 WMMA ----
#pragma unroll
        for (int kc = 0; kc < 2; ++kc) {
            const int m_lane = lane & 15;
            const int kbase  = (lane >> 4) * 16;   // A frag k-base within 32
            const int kbb    = (lane >> 4) * 16;   // B frag k-base within 32
            v16bf af[2], bfr[2];
#pragma unroll
            for (int mt = 0; mt < 2; ++mt) {
                const __bf16* ap = sA + (size_t)(wrow * 32 + mt * 16 + m_lane) * LDK + kc * 32;
#pragma unroll
                for (int e = 0; e < 16; ++e)
                    af[mt][e] = ap[kbase + e];
            }
#pragma unroll
            for (int nt = 0; nt < 2; ++nt) {
                const __bf16* wp = sW + (size_t)(wcol * 32 + nt * 16 + (lane & 15)) * LDK + kc * 32 + kbb;
#pragma unroll
                for (int e = 0; e < 16; ++e) bfr[nt][e] = wp[e];
            }
#pragma unroll
            for (int mt = 0; mt < 2; ++mt)
#pragma unroll
                for (int nt = 0; nt < 2; ++nt)
                    acc[mt][nt] = __builtin_amdgcn_wmma_f32_16x16x32_bf16(
                        false, af[mt], false, bfr[nt], (short)0, acc[mt][nt], false, false);
        }
        __syncthreads();
    }

    // ---- epilogue: C layout m=(lane>>4)*8+j, n=lane&15 ----
#pragma unroll
    for (int nt = 0; nt < 2; ++nt) {
        int n = n0 + wcol * 32 + nt * 16 + (lane & 15);
        if (n < N) {
            float bv = bias ? bias[n] : 0.0f;
#pragma unroll
            for (int mt = 0; mt < 2; ++mt) {
                int mbase = m0 + wrow * 32 + mt * 16 + ((lane >> 4) * 8);
#pragma unroll
                for (int j = 0; j < 8; ++j) {
                    float v = acc[mt][nt][j] + bv;
                    if (act == 1) v = sigf(v);
                    C[(size_t)(mbase + j) * ldc + n] = v;
                }
            }
        }
    }
}

// ---- f32 -> bf16 strided convert, 8 elems/thread (cols % 8 == 0) ----
__global__ void cvt_bf16(const float* __restrict__ src, __bf16* __restrict__ dst,
                         int rows, int cols, int src_ld, int dst_ld, int dst_off) {
    size_t i8 = (size_t)blockIdx.x * blockDim.x + threadIdx.x;
    int cpr = cols >> 3;
    size_t total8 = (size_t)rows * cpr;
    if (i8 >= total8) return;
    int r = (int)(i8 / cpr), c = (int)(i8 % cpr) * 8;
    const float* s = src + (size_t)r * src_ld + c;
    float4 f0 = *(const float4*)s;
    float4 f1 = *(const float4*)(s + 4);
    bf16x8 v;
    v[0] = (__bf16)f0.x; v[1] = (__bf16)f0.y; v[2] = (__bf16)f0.z; v[3] = (__bf16)f0.w;
    v[4] = (__bf16)f1.x; v[5] = (__bf16)f1.y; v[6] = (__bf16)f1.z; v[7] = (__bf16)f1.w;
    *(bf16x8*)(dst + (size_t)r * dst_ld + dst_off + c) = v;
}

__global__ void addvec(const float* __restrict__ a, const float* __restrict__ b,
                       float* __restrict__ o, int n) {
    int i = blockIdx.x * blockDim.x + threadIdx.x;
    if (i < n) o[i] = a[i] + b[i];
}

// ---- embedding gather: embT[b,t,:] = table[captions[b,t],:] ----
__global__ void emb_gather(const float* __restrict__ table, const int* __restrict__ caps,
                           float* __restrict__ embT) {
    int bt = blockIdx.x;                // 0 .. B*T-1
    int tok = caps[bt];
    if (tok < 0) tok = 0;
    if (tok >= Vn) tok = Vn - 1;
    const float* src = table + (size_t)tok * EMBn;
    float* dst = embT + (size_t)bt * EMBn;
    for (int i = threadIdx.x; i < EMBn; i += blockDim.x) dst[i] = src[i];
}

// ---- mean over P: mean[b,e] ----
__global__ void mean_enc_k(const float* __restrict__ enc, float* __restrict__ mean) {
    int idx = blockIdx.x * blockDim.x + threadIdx.x;     // B*E
    if (idx >= Bn * En) return;
    int b = idx / En, e = idx % En;
    const float* p = enc + (size_t)b * Pn * En + e;
    float s = 0.0f;
    for (int i = 0; i < Pn; ++i) s += p[(size_t)i * En];
    mean[idx] = s * (1.0f / (float)Pn);
}

// ---- fused attention step: e = tanh(att1+att2)·Wfull ; softmax ; context ----
__global__ void attention_step(const float* __restrict__ att1,  // [B*P, A]
                               const float* __restrict__ att2,  // [B, A]
                               const float* __restrict__ Wfull, // [A]
                               const float* __restrict__ bfull, // [1]
                               const float* __restrict__ enc,   // [B,P,E]
                               float* __restrict__ context,     // [B,E]
                               float* __restrict__ alphas,      // [B,T,P]
                               int t) {
    __shared__ float s_att2[An];
    __shared__ float s_wf[An];
    __shared__ float s_e[256];
    __shared__ float s_red[8];
    __shared__ float s_scl[2];

    int b = blockIdx.x, tid = threadIdx.x;
    int lane = tid & 31, wave = tid >> 5;

    for (int i = tid; i < An; i += 256) {
        s_att2[i] = att2[(size_t)b * An + i];
        s_wf[i]   = Wfull[i];
    }
    if (tid >= Pn && tid < 256) s_e[tid] = -3.0e38f;
    __syncthreads();

    // e[p], one p per wave at a time
    for (int p = wave; p < Pn; p += 8) {
        const float* a1 = att1 + ((size_t)(b * Pn + p)) * An;
        float s = 0.0f;
        for (int a = lane; a < An; a += 32)
            s += tanhf(a1[a] + s_att2[a]) * s_wf[a];
        for (int off = 16; off; off >>= 1) s += __shfl_xor(s, off, 32);
        if (lane == 0) s_e[p] = s + bfull[0];
    }
    __syncthreads();

    // softmax over 196 entries (padded with -inf)
    float v = s_e[tid];
    float m = v;
    for (int off = 16; off; off >>= 1) m = fmaxf(m, __shfl_xor(m, off, 32));
    if (lane == 0) s_red[wave] = m;
    __syncthreads();
    if (tid == 0) {
        float mm = s_red[0];
        for (int i = 1; i < 8; ++i) mm = fmaxf(mm, s_red[i]);
        s_scl[0] = mm;
    }
    __syncthreads();
    float ev = (tid < Pn) ? __expf(v - s_scl[0]) : 0.0f;
    float s = ev;
    for (int off = 16; off; off >>= 1) s += __shfl_xor(s, off, 32);
    __syncthreads();
    if (lane == 0) s_red[wave] = s;
    __syncthreads();
    if (tid == 0) {
        float ss = 0.0f;
        for (int i = 0; i < 8; ++i) ss += s_red[i];
        s_scl[1] = ss;
    }
    __syncthreads();
    float alpha = ev / s_scl[1];
    s_e[tid] = alpha;
    if (tid < Pn) alphas[((size_t)b * Tn + t) * Pn + tid] = alpha;
    __syncthreads();

    // context[b,e] = sum_p alpha[p] * enc[b,p,e]
    for (int e0 = tid; e0 < En; e0 += 256) {
        const float* ep = enc + (size_t)b * Pn * En + e0;
        float acc = 0.0f;
        for (int p = 0; p < Pn; ++p) acc += s_e[p] * ep[(size_t)p * En];
        context[(size_t)b * En + e0] = acc;
    }
}

// ---- build x2 = [emb_t (512) | gate*context (2048) | h (512)] ----
__global__ void build_x2(const float* __restrict__ embT, const float* __restrict__ gate,
                         const float* __restrict__ context, const float* __restrict__ h,
                         float* __restrict__ x2, int t) {
    int b = blockIdx.x;
    for (int i = threadIdx.x; i < XW; i += blockDim.x) {
        float v;
        if (i < EMBn)           v = embT[((size_t)b * Tn + t) * EMBn + i];
        else if (i < EMBn + En) { int j = i - EMBn; v = gate[(size_t)b * En + j] * context[(size_t)b * En + j]; }
        else                    v = h[(size_t)b * Dn + (i - EMBn - En)];
        x2[(size_t)b * XW + i] = v;
    }
}

// ---- LSTM pointwise: gate order i,f,g,o ----
__global__ void lstm_point(const float* __restrict__ gates,
                           float* __restrict__ h, float* __restrict__ c) {
    int idx = blockIdx.x * blockDim.x + threadIdx.x;     // B*D
    if (idx >= Bn * Dn) return;
    int b = idx >> 9, d = idx & (Dn - 1);
    const float* g = gates + (size_t)b * 2048;
    float i_ = sigf(g[d]);
    float f_ = sigf(g[Dn + d]);
    float gg = tanhf(g[2 * Dn + d]);
    float o_ = sigf(g[3 * Dn + d]);
    float cn = f_ * c[idx] + i_ * gg;
    float hn = o_ * tanhf(cn);
    c[idx] = cn;
    h[idx] = hn;
}

// ===================================================================
extern "C" void kernel_launch(void* const* d_in, const int* in_sizes, int n_in,
                              void* d_out, int out_size, void* d_ws, size_t ws_size,
                              hipStream_t stream) {
    const float* enc      = (const float*)d_in[0];
    const int*   captions = (const int*)  d_in[1];
    // d_in[2] lengths: unused by reference compute
    const float* W_enc    = (const float*)d_in[3];
    const float* b_enc    = (const float*)d_in[4];
    const float* W_dec    = (const float*)d_in[5];
    const float* b_dec    = (const float*)d_in[6];
    const float* W_full   = (const float*)d_in[7];
    const float* b_full   = (const float*)d_in[8];
    const float* emb_tab  = (const float*)d_in[9];
    const float* W_ih     = (const float*)d_in[10];
    const float* b_ih     = (const float*)d_in[11];
    const float* W_hh     = (const float*)d_in[12];
    const float* b_hh     = (const float*)d_in[13];
    const float* W_init_h = (const float*)d_in[14];
    const float* b_init_h = (const float*)d_in[15];
    const float* W_init_c = (const float*)d_in[16];
    const float* b_init_c = (const float*)d_in[17];
    const float* W_beta   = (const float*)d_in[18];
    const float* b_beta   = (const float*)d_in[19];
    const float* W_fc     = (const float*)d_in[20];
    const float* b_fc     = (const float*)d_in[21];

    float* preds  = (float*)d_out;                               // [B,T,V]
    float* alphas = (float*)d_out + (size_t)Bn * Tn * Vn;        // [B,T,P]

    // ---------------- workspace carve-up ----------------
    char* ws = (char*)d_ws;
    size_t off = 0;
    auto carve = [&](size_t bytes) -> char* {
        off = (off + 255) & ~(size_t)255;
        char* p = ws + off;
        off += bytes;
        return p;
    };
    float*  att1    = (float*) carve((size_t)Bn * Pn * An * 4);
    __bf16* WencB   = (__bf16*)carve((size_t)An * En * 2);
    __bf16* WdecB   = (__bf16*)carve((size_t)An * Dn * 2);
    __bf16* WbetaB  = (__bf16*)carve((size_t)En * Dn * 2);
    __bf16* WcatB   = (__bf16*)carve((size_t)2048 * XW * 2);
    __bf16* WfcB    = (__bf16*)carve((size_t)NFCPAD * Dn * 2);   // padded rows
    __bf16* WinitHB = (__bf16*)carve((size_t)Dn * En * 2);
    __bf16* WinitCB = (__bf16*)carve((size_t)Dn * En * 2);
    float*  bcat    = (float*) carve(2048 * 4);
    float*  embT    = (float*) carve((size_t)Bn * Tn * EMBn * 4);
    float*  meanb   = (float*) carve((size_t)Bn * En * 4);
    float*  hbuf    = (float*) carve((size_t)Bn * Dn * 4);
    float*  cbuf    = (float*) carve((size_t)Bn * Dn * 4);
    float*  att2b   = (float*) carve((size_t)Bn * An * 4);
    float*  gateb   = (float*) carve((size_t)Bn * En * 4);
    float*  ctxb    = (float*) carve((size_t)Bn * En * 4);
    float*  x2b     = (float*) carve((size_t)Bn * XW * 4);
    float*  gatesb  = (float*) carve((size_t)Bn * 2048 * 4);
    (void)ws_size; (void)in_sizes; (void)n_in; (void)out_size;

    auto cvt = [&](const float* s, __bf16* d, int r, int c, int sld, int dld, int doff) {
        size_t tot8 = ((size_t)r * c) >> 3;
        cvt_bf16<<<dim3((unsigned)((tot8 + 255) / 256)), dim3(256), 0, stream>>>(s, d, r, c, sld, dld, doff);
    };

    // -------- one-time prep --------
    cvt(W_enc,    WencB,   An,   En, En, En, 0);
    cvt(W_dec,    WdecB,   An,   Dn, Dn, Dn, 0);
    cvt(W_beta,   WbetaB,  En,   Dn, Dn, Dn, 0);
    cvt(W_ih,     WcatB,   2048, EMBn + En, EMBn + En, XW, 0);       // cols 0..2559
    cvt(W_hh,     WcatB,   2048, Dn, Dn, XW, EMBn + En);             // cols 2560..3071
    cvt(W_fc,     WfcB,    Vn,   Dn, Dn, Dn, 0);                     // rows 10000..10111 stay garbage (discarded)
    cvt(W_init_h, WinitHB, Dn,   En, En, En, 0);
    cvt(W_init_c, WinitCB, Dn,   En, En, En, 0);
    addvec<<<dim3(8), dim3(256), 0, stream>>>(b_ih, b_hh, bcat, 2048);

    emb_gather<<<dim3(Bn * Tn), dim3(256), 0, stream>>>(emb_tab, captions, embT);
    mean_enc_k<<<dim3((Bn * En + 255) / 256), dim3(256), 0, stream>>>(enc, meanb);

    auto gemm = [&](const float* A, const __bf16* W, const float* bias, float* C,
                    int M, int N, int K, int ldc, int act) {
        dim3 grid((N + NT - 1) / NT, M / MT);
        gemm_bf16_wmma<<<grid, dim3(256), 0, stream>>>(A, W, bias, C, M, N, K, ldc, act);
    };

    // att1 = enc @ W_enc^T + b_enc   [B*P, A]
    gemm(enc, WencB, b_enc, att1, Bn * Pn, An, En, An, 0);
    // h0 / c0 from mean_enc
    gemm(meanb, WinitHB, b_init_h, hbuf, Bn, Dn, En, Dn, 0);
    gemm(meanb, WinitCB, b_init_c, cbuf, Bn, Dn, En, Dn, 0);

    // -------- recurrence --------
    for (int t = 0; t < Tn; ++t) {
        // att2 = h @ W_dec^T + b_dec
        gemm(hbuf, WdecB, b_dec, att2b, Bn, An, Dn, An, 0);
        // e/softmax/context
        attention_step<<<dim3(Bn), dim3(256), 0, stream>>>(
            att1, att2b, W_full, b_full, enc, ctxb, alphas, t);
        // beta gate = sigmoid(h @ W_beta^T + b_beta)
        gemm(hbuf, WbetaB, b_beta, gateb, Bn, En, Dn, En, 1);
        // x2 = [emb_t | gate*ctx | h]
        build_x2<<<dim3(Bn), dim3(256), 0, stream>>>(embT, gateb, ctxb, hbuf, x2b, t);
        // gates = x2 @ Wcat^T + (b_ih + b_hh)
        gemm(x2b, WcatB, bcat, gatesb, Bn, 2048, XW, 2048, 0);
        // LSTM pointwise update of h, c
        lstm_point<<<dim3((Bn * Dn + 255) / 256), dim3(256), 0, stream>>>(gatesb, hbuf, cbuf);
        // preds[:, t, :] = h_new @ W_fc^T + b_fc   (row stride T*V)
        gemm(hbuf, WfcB, b_fc, preds + (size_t)t * Vn, Bn, Vn, Dn, Tn * Vn, 0);
    }
}
